// AttentionModule_53223234732422
// MI455X (gfx1250) — matrix operand order repaired and checked by reading, not compiled, not run on hardware
//
#include <hip/hip_runtime.h>
#include <math.h>

#define S_TOK 2048
#define EMB   2048
#define NH    16
#define HD    128

typedef __attribute__((ext_vector_type(16))) __bf16       v16bf;
typedef __attribute__((ext_vector_type(8)))  float        v8f;
typedef __attribute__((ext_vector_type(4)))  unsigned int v4u;
typedef __attribute__((ext_vector_type(4)))  float        v4f;

union Frag { v16bf v; v4u q[2]; };

__device__ __forceinline__ unsigned short f2bf(float f) {
  union { float f; unsigned u; } c; c.f = f;
  unsigned r = (c.u + 0x7FFFu + ((c.u >> 16) & 1u)) >> 16;   // round-to-nearest-even
  return (unsigned short)r;
}
__device__ __forceinline__ float bf2f(unsigned short b) {
  union { unsigned u; float f; } c; c.u = ((unsigned)b) << 16;
  return c.f;
}
__device__ __forceinline__ v8f wmma_bf16(v16bf a, v16bf b, v8f c) {
  return __builtin_amdgcn_wmma_f32_16x16x32_bf16(false, a, false, b, (short)0, c,
                                                 false, false);
}
// CDNA5 async copy: global -> LDS, 16B per lane, tracked by ASYNCcnt.
__device__ __forceinline__ void async_ld16(unsigned lds, const void* g) {
  asm volatile("global_load_async_to_lds_b128 %0, %1, off"
               :: "v"(lds), "v"(g) : "memory");
}

// ---------------------------------------------------------------- RMSNorm -> bf16
__global__ __launch_bounds__(256) void rmsnorm_kernel(const float* __restrict__ xs,
                                                      const float* __restrict__ w,
                                                      unsigned short* __restrict__ xn) {
  __shared__ float red[256];
  const int row = blockIdx.x, t = threadIdx.x;
  const float* x = xs + (size_t)row * EMB;
  float ss = 0.f;
  for (int j = t; j < EMB; j += 256) { float v = x[j]; ss += v * v; }
  red[t] = ss;
  __syncthreads();
  for (int s = 128; s > 0; s >>= 1) {
    if (t < s) red[t] += red[t + s];
    __syncthreads();
  }
  const float inv = rsqrtf(red[0] * (1.f / EMB) + 1e-6f);
  unsigned short* o = xn + (size_t)row * EMB;
  for (int j = t; j < EMB; j += 256) o[j] = f2bf(x[j] * inv * w[j]);
}

// ---------------------------------------------------------------- fp32 -> bf16 weights
__global__ __launch_bounds__(256) void cvtw_kernel(const float* __restrict__ w,
                                                   unsigned short* __restrict__ wb) {
  const int i = (blockIdx.x * 256 + threadIdx.x) * 8;
  v4f a = *(const v4f*)&w[i];
  v4f b = *(const v4f*)&w[i + 4];
  v4u o;
  o[0] = (unsigned)f2bf(a[0]) | ((unsigned)f2bf(a[1]) << 16);
  o[1] = (unsigned)f2bf(a[2]) | ((unsigned)f2bf(a[3]) << 16);
  o[2] = (unsigned)f2bf(b[0]) | ((unsigned)f2bf(b[1]) << 16);
  o[3] = (unsigned)f2bf(b[2]) | ((unsigned)f2bf(b[3]) << 16);
  *(v4u*)&wb[i] = o;
}

// ---------------------------------------------------------------- RoPE (in place, bf16)
__global__ __launch_bounds__(256) void rope_kernel(unsigned short* __restrict__ buf) {
  const int row = blockIdx.x, t = threadIdx.x;
  for (int i = 0; i < 4; ++i) {
    const int p = t + i * 256;          // pair index within row, 0..1023
    const int ip = p & 63;              // pair index within head (half = 64)
    const float theta = __powf(10000.f, -(float)ip * (1.f / 64.f));
    const float ang = (float)row * theta;
    float sn, cs;
    __sincosf(ang, &sn, &cs);
    const size_t off = (size_t)row * EMB + 2 * p;
    const float x0 = bf2f(buf[off]), x1 = bf2f(buf[off + 1]);
    buf[off]     = f2bf(cs * x0 - sn * x1);
    buf[off + 1] = f2bf(sn * x0 + cs * x1);
  }
}

// ---------------------------------------------------------------- tiled WMMA GEMM
// C[S,E] = A(bf16)[S,E] @ W(bf16)[E,E]^T.  mode 0: bf16 out, 1: bf16 transposed out,
// 2: fp32 out.  Block tile 128x128, BK=32, 8 waves (4x2), wave tile 32x64.
// Double-buffered LDS filled with global_load_async_to_lds_b128 (ASYNCcnt), so the
// copy of tile k+1 overlaps the WMMA work on tile k with no VGPR staging.
__global__ __launch_bounds__(256) void gemm_kernel(const unsigned short* __restrict__ A,
                                                   const unsigned short* __restrict__ Bw,
                                                   void* __restrict__ Cout, int mode) {
  __shared__ alignas(16) unsigned short As[2][128 * 40];   // [row][k], +8 pad
  __shared__ alignas(16) unsigned short Bs[2][128 * 40];   // [out-col][k], +8 pad
  const int t = threadIdx.x;
  const int wave = t >> 5, lane = t & 31;
  const int wy = wave >> 1, wx = wave & 1;
  const int m0 = lane & 15, kh = lane >> 4;
  const int rowBlk = blockIdx.y * 128, colBlk = blockIdx.x * 128;

  // per-thread copy slots: 2 x 16B chunks for A, 2 for B (4 async instrs per wave)
  const int r0 = t >> 2,          c0 = (t & 3) * 8;
  const int r1 = (t + 256) >> 2,  c1 = ((t + 256) & 3) * 8;
  const unsigned short* gA0 = &A [(size_t)(rowBlk + r0) * EMB + c0];
  const unsigned short* gA1 = &A [(size_t)(rowBlk + r1) * EMB + c1];
  const unsigned short* gB0 = &Bw[(size_t)(colBlk + r0) * EMB + c0];
  const unsigned short* gB1 = &Bw[(size_t)(colBlk + r1) * EMB + c1];
  const unsigned ldsA0 = (unsigned)(size_t)&As[0][r0 * 40 + c0];
  const unsigned ldsA1 = (unsigned)(size_t)&As[0][r1 * 40 + c1];
  const unsigned ldsB0 = (unsigned)(size_t)&Bs[0][r0 * 40 + c0];
  const unsigned ldsB1 = (unsigned)(size_t)&Bs[0][r1 * 40 + c1];
  const unsigned BUFB = 128 * 40 * 2;                      // bytes per LDS buffer

  v8f acc[2][4];
  for (int rt = 0; rt < 2; ++rt)
    for (int ct = 0; ct < 4; ++ct)
      for (int j = 0; j < 8; ++j) acc[rt][ct][j] = 0.f;

  // prologue: fill buffer 0 with tile kb=0
  async_ld16(ldsA0, gA0);
  async_ld16(ldsA1, gA1);
  async_ld16(ldsB0, gB0);
  async_ld16(ldsB1, gB1);

  unsigned cur = 0;
  for (int kb = 0; kb < EMB; kb += 32) {
    if (kb + 32 < EMB) {                                   // issue next tile -> other buf
      const unsigned off = (cur ^ 1u) * BUFB;
      async_ld16(ldsA0 + off, gA0 + kb + 32);
      async_ld16(ldsA1 + off, gA1 + kb + 32);
      async_ld16(ldsB0 + off, gB0 + kb + 32);
      async_ld16(ldsB1 + off, gB1 + kb + 32);
      asm volatile("s_wait_asynccnt 4" ::: "memory");      // prev group (cur buf) done
    } else {
      asm volatile("s_wait_asynccnt 0" ::: "memory");
    }
    __syncthreads();

    const unsigned short* Ac = As[cur];
    const unsigned short* Bc = Bs[cur];
    Frag a[2];
    for (int rt = 0; rt < 2; ++rt) {
      const int row = wy * 32 + rt * 16 + m0;
      a[rt].q[0] = *(const v4u*)&Ac[row * 40 + kh * 8];    // K = kh*8 .. +8
      a[rt].q[1] = *(const v4u*)&Ac[row * 40 + 16 + kh * 8];
    }
    for (int ct = 0; ct < 4; ++ct) {
      Frag b;
      const int col = wx * 64 + ct * 16 + m0;
      b.q[0] = *(const v4u*)&Bc[col * 40 + kh * 8];        // K = kh*8 .. +8
      b.q[1] = *(const v4u*)&Bc[col * 40 + 16 + kh * 8];
      for (int rt = 0; rt < 2; ++rt)
        acc[rt][ct] = wmma_bf16(a[rt].v, b.v, acc[rt][ct]);
    }
    __syncthreads();                                       // all waves done with cur
    cur ^= 1u;
  }

  for (int rt = 0; rt < 2; ++rt)
    for (int ct = 0; ct < 4; ++ct)
      for (int j = 0; j < 8; ++j) {
        const int row = rowBlk + wy * 32 + rt * 16 + j + 8 * kh;
        const int col = colBlk + wx * 64 + ct * 16 + m0;
        const float val = acc[rt][ct][j];
        if (mode == 0)      ((unsigned short*)Cout)[(size_t)row * EMB + col]   = f2bf(val);
        else if (mode == 1) ((unsigned short*)Cout)[(size_t)col * S_TOK + row] = f2bf(val);
        else                ((float*)Cout)[(size_t)row * EMB + col]            = val;
      }
}

// ---------------------------------------------------------------- flash attention
// One wave = 16 query rows of one head; online softmax; Q*K^T and P*V via bf16 WMMA.
__global__ __launch_bounds__(256) void attn_kernel(const unsigned short* __restrict__ Q,
                                                   const unsigned short* __restrict__ K,
                                                   const unsigned short* __restrict__ Vt,
                                                   unsigned short* __restrict__ O) {
  __shared__ alignas(16) unsigned short Pst[8 * 16 * 40];  // per-wave P staging
  const int t = threadIdx.x, wave = t >> 5, lane = t & 31;
  const int m0 = lane & 15, kh = lane >> 4;
  const int h = blockIdx.x;
  const int qbase = blockIdx.y * 128 + wave * 16;
  const int hoff = h * HD;
  unsigned short* P = &Pst[wave * 16 * 40];

  Frag qf[4];                                    // Q rows, full head dim (4 x K=32)
  for (int f = 0; f < 4; ++f) {
    const unsigned short* p = &Q[(size_t)(qbase + m0) * EMB + hoff + f * 32];
    qf[f].q[0] = *(const v4u*)&p[kh * 8];
    qf[f].q[1] = *(const v4u*)&p[16 + kh * 8];
  }

  float mst[8], lst[8];
  v8f o[8];
  for (int j = 0; j < 8; ++j) { mst[j] = -__builtin_inff(); lst[j] = 0.f; }
  for (int tt = 0; tt < 8; ++tt)
    for (int j = 0; j < 8; ++j) o[tt][j] = 0.f;

  const float sc = 0.0883883476483f;             // 1/sqrt(128)

  for (int c = 0; c < qbase + 16; c += 32) {     // causal KV range, 32 keys/step
    v8f s0, s1;
    for (int j = 0; j < 8; ++j) { s0[j] = 0.f; s1[j] = 0.f; }
    for (int f = 0; f < 4; ++f) {                // K-dim 128 in 4 x 32
      Frag b0, b1;
      const unsigned short* k0 = &K[(size_t)(c + m0) * EMB + hoff + f * 32];
      const unsigned short* k1 = &K[(size_t)(c + 16 + m0) * EMB + hoff + f * 32];
      b0.q[0] = *(const v4u*)(k0 + kh * 8); b0.q[1] = *(const v4u*)(k0 + 16 + kh * 8);
      b1.q[0] = *(const v4u*)(k1 + kh * 8); b1.q[1] = *(const v4u*)(k1 + 16 + kh * 8);
      s0 = wmma_bf16(qf[f].v, b0.v, s0);
      s1 = wmma_bf16(qf[f].v, b1.v, s1);
    }
    for (int j = 0; j < 8; ++j) {
      const int qr = qbase + j + 8 * kh;         // this lane-half's query row
      float v0 = (c + m0      <= qr) ? s0[j] * sc : -__builtin_inff();
      float v1 = (c + 16 + m0 <= qr) ? s1[j] * sc : -__builtin_inff();
      float mx = fmaxf(v0, v1);
      for (int d = 1; d < 16; d <<= 1) mx = fmaxf(mx, __shfl_xor(mx, d, 32));
      const float mn = fmaxf(mst[j], mx);
      const float al = __expf(mst[j] - mn);
      const float p0 = __expf(v0 - mn);
      const float p1 = __expf(v1 - mn);
      float rs = p0 + p1;
      for (int d = 1; d < 16; d <<= 1) rs += __shfl_xor(rs, d, 32);
      lst[j] = lst[j] * al + rs;
      mst[j] = mn;
      for (int tt = 0; tt < 8; ++tt) o[tt][j] *= al;
      P[(j + 8 * kh) * 40 + m0]      = f2bf(p0); // D-layout -> LDS (row, col)
      P[(j + 8 * kh) * 40 + 16 + m0] = f2bf(p1);
    }
    asm volatile("" ::: "memory");               // order LDS store -> load at IR level
    Frag pf;                                     // reload as A-layout 16x32
    pf.q[0] = *(const v4u*)&P[m0 * 40 + kh * 8];
    pf.q[1] = *(const v4u*)&P[m0 * 40 + 16 + kh * 8];
    for (int tt = 0; tt < 8; ++tt) {             // P(16x32) x V(32x16) over 8 dim tiles
      Frag vf;
      const unsigned short* vp = &Vt[(size_t)(hoff + tt * 16 + m0) * S_TOK + c];
      vf.q[0] = *(const v4u*)(vp + kh * 8); vf.q[1] = *(const v4u*)(vp + 16 + kh * 8);
      o[tt] = wmma_bf16(pf.v, vf.v, o[tt]);
    }
    asm volatile("" ::: "memory");               // WAR vs next-iter P stores
  }

  for (int j = 0; j < 8; ++j) {
    const float inv = 1.0f / lst[j];
    const size_t row = (size_t)(qbase + j + 8 * kh);
    for (int tt = 0; tt < 8; ++tt)
      O[row * EMB + hoff + tt * 16 + m0] = f2bf(o[tt][j] * inv);
  }
}

// ---------------------------------------------------------------- launch
extern "C" void kernel_launch(void* const* d_in, const int* in_sizes, int n_in,
                              void* d_out, int out_size, void* d_ws, size_t ws_size,
                              hipStream_t stream) {
  (void)in_sizes; (void)n_in; (void)out_size; (void)ws_size;
  const float* xs = (const float*)d_in[0];
  const float* nw = (const float*)d_in[1];
  const float* wq = (const float*)d_in[2];
  const float* wk = (const float*)d_in[3];
  const float* wv = (const float*)d_in[4];
  const float* wo = (const float*)d_in[5];
  float* out = (float*)d_out;

  char* ws = (char*)d_ws;
  const size_t MB = 1u << 20;
  unsigned short* XN = (unsigned short*)(ws);            // bf16 [S,E]
  unsigned short* QB = (unsigned short*)(ws + 8 * MB);   // bf16 [S,E]
  unsigned short* KB = (unsigned short*)(ws + 16 * MB);  // bf16 [S,E]
  unsigned short* VT = (unsigned short*)(ws + 24 * MB);  // bf16 [E,S] (transposed V)
  unsigned short* OB = (unsigned short*)(ws + 32 * MB);  // bf16 [S,E]
  unsigned short* WB = (unsigned short*)(ws + 40 * MB);  // bf16 weight staging [E,E]

  const dim3 gGemm(EMB / 128, S_TOK / 128);
  const dim3 gAttn(NH, S_TOK / 128);

  rmsnorm_kernel<<<S_TOK, 256, 0, stream>>>(xs, nw, XN);

  cvtw_kernel<<<EMB * EMB / (256 * 8), 256, 0, stream>>>(wq, WB);
  gemm_kernel<<<gGemm, 256, 0, stream>>>(XN, WB, QB, 0);
  rope_kernel<<<S_TOK, 256, 0, stream>>>(QB);

  cvtw_kernel<<<EMB * EMB / (256 * 8), 256, 0, stream>>>(wk, WB);
  gemm_kernel<<<gGemm, 256, 0, stream>>>(XN, WB, KB, 0);
  rope_kernel<<<S_TOK, 256, 0, stream>>>(KB);

  cvtw_kernel<<<EMB * EMB / (256 * 8), 256, 0, stream>>>(wv, WB);
  gemm_kernel<<<gGemm, 256, 0, stream>>>(XN, WB, VT, 1);

  attn_kernel<<<gAttn, 256, 0, stream>>>(QB, KB, VT, OB);

  cvtw_kernel<<<EMB * EMB / (256 * 8), 256, 0, stream>>>(wo, WB);
  gemm_kernel<<<gGemm, 256, 0, stream>>>(OB, WB, out, 2);
}
